// LlamaAttention_41162966565419
// MI455X (gfx1250) — compile-verified, matrix operand order checked by reading, not run on hardware
//
#include <hip/hip_runtime.h>
#include <hip/hip_bf16.h>
#include <stdint.h>

#define H_HEADS 16
#define KV_HEADS 4
#define HD 128
#define DMODEL 2048
#define DKV 512

typedef __attribute__((ext_vector_type(16))) __bf16 v16bf;
typedef __attribute__((ext_vector_type(8))) float v8f;

union FragB16 { v16bf v; uint32_t u[8]; };

__device__ __forceinline__ uint16_t f2bf(float f) {
  uint32_t u = __float_as_uint(f);
  u += 0x7FFFu + ((u >> 16) & 1u);
  return (uint16_t)(u >> 16);
}

// Build a 16x32 bf16 A/B fragment from LDS stored row-major [16 rows][>=koff+32 cols].
// ISA 7.12.2: lane m = L&15 holds row m; lanes 0-15 hold K 0-7/16-23, lanes 16-31 hold
// K 8-15/24-31, packed two bf16 per VGPR.
__device__ __forceinline__ void load_frag_lds(FragB16& f, const uint16_t* base, int stride,
                                              int koff, int lane) {
  const int r = lane & 15;
  const int hi = (lane >> 4) & 1;
  const uint16_t* row = base + r * stride + koff;
#pragma unroll
  for (int p = 0; p < 8; ++p) {
    const int kb = (p < 4) ? (hi * 8 + 2 * p) : (16 + hi * 8 + 2 * (p - 4));
    f.u[p] = *(const uint32_t*)(row + kb);
  }
}

__device__ __forceinline__ v8f wmma_bf16(const FragB16& a, const FragB16& b, v8f c) {
  return __builtin_amdgcn_wmma_f32_16x16x32_bf16(false, a.v, false, b.v, (short)0, c, false, false);
}

__device__ __forceinline__ void async_lds_b128(uint32_t lds_off, const void* gaddr) {
  asm volatile("global_load_async_to_lds_b128 %0, %1, off"
               :: "v"(lds_off), "v"((uint64_t)(uintptr_t)gaddr) : "memory");
}
__device__ __forceinline__ void wait_async0() {
  asm volatile("s_wait_asynccnt 0" ::: "memory");
}

// ---------------------------------------------------------------------------
// One-time f32 -> bf16 pack (vectorized x4).
// ---------------------------------------------------------------------------
__global__ __launch_bounds__(256) void pack_bf16(const float* __restrict__ src,
                                                 uint16_t* __restrict__ dst, int n4) {
  const int i = blockIdx.x * 256 + threadIdx.x;
  if (i >= n4) return;
  const float4 f = ((const float4*)src)[i];
  ushort4 o4;
  o4.x = f2bf(f.x); o4.y = f2bf(f.y); o4.z = f2bf(f.z); o4.w = f2bf(f.w);
  *(ushort4*)(dst + (size_t)i * 4) = o4;
}

// ---------------------------------------------------------------------------
// C[m][n] = sum_k A[m][k] * W[n][k]   (A: [M,K] bf16, W: [N,K] bf16, C: [M,N] f32)
// Block: 256 threads (8 waves), tile 128(M) x 64(N), K-step 64. Grid (N/64, M/128).
// Tiles staged with async-to-LDS DMA (no VALU, no VGPR round trip).
// ---------------------------------------------------------------------------
#define GS 72  // LDS row stride in bf16: 144B rows, 16B aligned, 36-bank skew
__global__ __launch_bounds__(256) void gemm_abt_bf16(const uint16_t* __restrict__ A,
                                                     const uint16_t* __restrict__ W,
                                                     float* __restrict__ C,
                                                     int M, int N, int K) {
  __shared__ __attribute__((aligned(16))) uint16_t sA[128 * GS];
  __shared__ __attribute__((aligned(16))) uint16_t sB[64 * GS];
  const int tid = threadIdx.x;
  const int lane = tid & 31, wave = tid >> 5;
  const int m0 = blockIdx.y * 128;
  const int n0 = blockIdx.x * 64;
  const uint32_t sAoff = (uint32_t)(uintptr_t)&sA[0];
  const uint32_t sBoff = (uint32_t)(uintptr_t)&sB[0];

  v8f acc[4];
#pragma unroll
  for (int nt = 0; nt < 4; ++nt)
#pragma unroll
    for (int r = 0; r < 8; ++r) acc[nt][r] = 0.0f;

  const int ar = tid >> 1;            // A: 128 rows x 8 16B-chunks
  const int ac0 = (tid & 1) * 4;
  const int br = tid >> 2;            // B: 64 rows x 8 16B-chunks
  const int bc0 = (tid & 3) * 2;

  for (int k0 = 0; k0 < K; k0 += 64) {
    __syncthreads();
#pragma unroll
    for (int it = 0; it < 4; ++it) {
      const int ch = ac0 + it;
      async_lds_b128(sAoff + (uint32_t)(ar * GS + ch * 8) * 2u,
                     A + (size_t)(m0 + ar) * K + k0 + ch * 8);
    }
#pragma unroll
    for (int it = 0; it < 2; ++it) {
      const int ch = bc0 + it;
      async_lds_b128(sBoff + (uint32_t)(br * GS + ch * 8) * 2u,
                     W + (size_t)(n0 + br) * K + k0 + ch * 8);
    }
    if (k0 + 64 < K) {  // hint next tiles toward L2 (global_prefetch_b8)
      __builtin_prefetch(A + (size_t)(m0 + ar) * K + k0 + 64 + ac0 * 8, 0, 0);
      __builtin_prefetch(W + (size_t)(n0 + br) * K + k0 + 64 + bc0 * 8, 0, 0);
    }
    wait_async0();
    __syncthreads();

    FragB16 a0, a1;
    load_frag_lds(a0, sA + wave * 16 * GS, GS, 0, lane);
    load_frag_lds(a1, sA + wave * 16 * GS, GS, 32, lane);
#pragma unroll
    for (int nt = 0; nt < 4; ++nt) {
      FragB16 b;
      load_frag_lds(b, sB + nt * 16 * GS, GS, 0, lane);
      acc[nt] = wmma_bf16(a0, b, acc[nt]);
      load_frag_lds(b, sB + nt * 16 * GS, GS, 32, lane);
      acc[nt] = wmma_bf16(a1, b, acc[nt]);
    }
  }

  const int hi = (lane >> 4) & 1, nn = lane & 15;
#pragma unroll
  for (int nt = 0; nt < 4; ++nt)
#pragma unroll
    for (int r = 0; r < 8; ++r) {
      const int row = m0 + wave * 16 + r + 8 * hi;
      C[(size_t)row * N + n0 + nt * 16 + nn] = acc[nt][r];
    }
}

// ---------------------------------------------------------------------------
// In-place rotary embedding on Q [S, 16*128] and K [S, 4*128], rotate-half.
// ---------------------------------------------------------------------------
__global__ __launch_bounds__(256) void rope_kernel(float* __restrict__ q, float* __restrict__ k,
                                                   const int* __restrict__ pos, int S) {
  const int idx = blockIdx.x * 256 + threadIdx.x;
  const int total = S * (H_HEADS + KV_HEADS) * (HD / 2);
  if (idx >= total) return;
  const int i = idx & 63;
  const int t = idx >> 6;
  const int head = t % (H_HEADS + KV_HEADS);
  const int s = t / (H_HEADS + KV_HEADS);
  const float p = (float)pos[s];
  const float invf = powf(10000.0f, -(float)(2 * i) / 128.0f);
  float sn, cs;
  sincosf(p * invf, &sn, &cs);
  float* base = (head < H_HEADS) ? (q + (size_t)s * DMODEL + head * HD)
                                 : (k + (size_t)s * DKV + (head - H_HEADS) * HD);
  const float x0 = base[i], x1 = base[i + 64];
  base[i]      = x0 * cs - x1 * sn;
  base[i + 64] = x1 * cs + x0 * sn;
}

// ---------------------------------------------------------------------------
// Pack to bf16 after RoPE:
//   qb  [S, 2048]  = q * (1/sqrt(128))       (softmax scale folded in)
//   kbf [S, 512]   = k
//   vtb [4*128, S] = v transposed            (per kv head, per dim, contiguous keys)
// ---------------------------------------------------------------------------
__global__ __launch_bounds__(256) void convert_post_rope(const float* __restrict__ q,
                                                         const float* __restrict__ k,
                                                         const float* __restrict__ v,
                                                         uint16_t* __restrict__ qb,
                                                         uint16_t* __restrict__ kbf,
                                                         uint16_t* __restrict__ vtb,
                                                         int S) {
  const size_t idx = (size_t)blockIdx.x * 256 + threadIdx.x;
  const size_t nq = (size_t)S * DMODEL;
  if (idx < nq) qb[idx] = f2bf(q[idx] * 0.08838834764831845f);
  const size_t nkv = (size_t)S * DKV;
  if (idx < nkv) {
    kbf[idx] = f2bf(k[idx]);
    const int s = (int)(idx / DKV);
    const int j = (int)(idx % DKV);   // j = kv*128 + d
    vtb[(size_t)j * S + s] = f2bf(v[idx]);
  }
}

// ---------------------------------------------------------------------------
// Flash attention: grid (S/128, H). 256 threads = 8 waves; wave owns 16 query rows.
// Causal, GQA (kv head = h/4). K/V^T tiles staged bf16 -> LDS with async-to-LDS DMA.
// Output written directly as bf16 for the O-projection GEMM.
// ---------------------------------------------------------------------------
#define SK_STRIDE 136   // 272B rows: 16B aligned for b128 async, 4-bank row skew
#define SV_STRIDE 72    // 144B rows: 16B aligned, 36-bank row skew
__global__ __launch_bounds__(256) void attn_kernel(const uint16_t* __restrict__ Qb,
                                                   const uint16_t* __restrict__ Kb,
                                                   const uint16_t* __restrict__ Vtb,
                                                   uint16_t* __restrict__ O,
                                                   int S) {
  __shared__ __attribute__((aligned(16))) uint16_t sK[64 * SK_STRIDE];   // [key][dim]
  __shared__ __attribute__((aligned(16))) uint16_t sVt[128 * SV_STRIDE]; // [dim][key]
  __shared__ uint16_t sP[8][16 * 66];                                    // per-wave P
  const int tid = threadIdx.x;
  const int lane = tid & 31, wave = tid >> 5;
  const int hi = (lane >> 4) & 1, nn = lane & 15;
  const int q0 = blockIdx.x * 128;
  const int h = blockIdx.y;
  const int kvh = h >> 2;
  const int qw0 = q0 + wave * 16;
  const float LOG2E = 1.4426950408889634f;

  // Q fragments: direct dword-pair loads from pre-scaled bf16 Q
  FragB16 qf[4];
  {
    const uint16_t* qrow = Qb + (size_t)(qw0 + nn) * DMODEL + h * HD;
#pragma unroll
    for (int ks = 0; ks < 4; ++ks)
#pragma unroll
      for (int p = 0; p < 8; ++p) {
        const int kb2 = ks * 32 + ((p < 4) ? (hi * 8 + 2 * p) : (16 + hi * 8 + 2 * (p - 4)));
        qf[ks].u[p] = *(const uint32_t*)(qrow + kb2);
      }
  }

  float mrow[8], lrow[8];
  v8f o[8];
#pragma unroll
  for (int r = 0; r < 8; ++r) { mrow[r] = -__builtin_inff(); lrow[r] = 0.0f; }
#pragma unroll
  for (int vt = 0; vt < 8; ++vt)
#pragma unroll
    for (int r = 0; r < 8; ++r) o[vt][r] = 0.0f;

  const uint32_t sKoff = (uint32_t)(uintptr_t)&sK[0];
  const uint32_t sVoff = (uint32_t)(uintptr_t)&sVt[0];

  for (int kb = 0; kb < q0 + 128; kb += 64) {
    __syncthreads();  // previous iteration's LDS reads complete
    {
      // K tile [64 keys x 128 dims] bf16: 16B async chunks (8 dims each)
      const int krow = tid >> 2;
      const int dg0 = (tid & 3) * 8;
#pragma unroll
      for (int it = 0; it < 4; ++it) {
        const int dg = dg0 + it * 32;
        async_lds_b128(sKoff + (uint32_t)(krow * SK_STRIDE + dg) * 2u,
                       Kb + (size_t)(kb + krow) * DKV + kvh * HD + dg);
      }
      // V^T tile [128 dims x 64 keys] bf16: contiguous keys per dim row
      const int d = tid >> 1;
      const int kc0 = (tid & 1) * 8;
#pragma unroll
      for (int it = 0; it < 4; ++it) {
        const int kc = kc0 + it * 16;
        async_lds_b128(sVoff + (uint32_t)(d * SV_STRIDE + kc) * 2u,
                       Vtb + (size_t)(kvh * HD + d) * S + kb + kc);
      }
      wait_async0();
    }
    __syncthreads();

    // scores: 16x64 per wave = 4 accumulators, K dim = 128 = 4 WMMA steps
    v8f sc[4];
#pragma unroll
    for (int nt = 0; nt < 4; ++nt) {
#pragma unroll
      for (int r = 0; r < 8; ++r) sc[nt][r] = 0.0f;
#pragma unroll
      for (int ks = 0; ks < 4; ++ks) {
        FragB16 b;
        load_frag_lds(b, sK + nt * 16 * SK_STRIDE, SK_STRIDE, ks * 32, lane);
        sc[nt] = wmma_bf16(qf[ks], b, sc[nt]);
      }
    }

    // causal mask (only boundary tiles need it)
    if (kb + 63 > qw0) {
#pragma unroll
      for (int nt = 0; nt < 4; ++nt)
#pragma unroll
        for (int r = 0; r < 8; ++r) {
          const int col = kb + nt * 16 + nn;
          const int rowg = qw0 + r + 8 * hi;
          if (col > rowg) sc[nt][r] = -__builtin_inff();
        }
    }

    // online softmax: row stats per (lane, r); reduce across 16-lane halves
    float scl[8];
#pragma unroll
    for (int r = 0; r < 8; ++r) {
      float v0 = fmaxf(fmaxf(sc[0][r], sc[1][r]), fmaxf(sc[2][r], sc[3][r]));
#pragma unroll
      for (int mk = 1; mk <= 8; mk <<= 1) v0 = fmaxf(v0, __shfl_xor(v0, mk, 32));
      const float mnew = fmaxf(mrow[r], v0);
      scl[r] = exp2f((mrow[r] - mnew) * LOG2E);
      mrow[r] = mnew;
    }
#pragma unroll
    for (int nt = 0; nt < 4; ++nt)
#pragma unroll
      for (int r = 0; r < 8; ++r)
        sc[nt][r] = exp2f((sc[nt][r] - mrow[r]) * LOG2E);
#pragma unroll
    for (int r = 0; r < 8; ++r) {
      float v0 = (sc[0][r] + sc[1][r]) + (sc[2][r] + sc[3][r]);
#pragma unroll
      for (int mk = 1; mk <= 8; mk <<= 1) v0 += __shfl_xor(v0, mk, 32);
      lrow[r] = lrow[r] * scl[r] + v0;
    }
#pragma unroll
    for (int vt = 0; vt < 8; ++vt)
#pragma unroll
      for (int r = 0; r < 8; ++r) o[vt][r] *= scl[r];

    // restage P (C-layout -> A-fragment layout) through per-wave LDS
    uint16_t* sPw = &sP[wave][0];
#pragma unroll
    for (int nt = 0; nt < 4; ++nt)
#pragma unroll
      for (int r = 0; r < 8; ++r)
        sPw[(r + 8 * hi) * 66 + nt * 16 + nn] = f2bf(sc[nt][r]);
    __syncthreads();

    // O += P @ V : P is 16x64 (2 K-steps), V^T gives B fragments over 8 dim-tiles
#pragma unroll
    for (int ks = 0; ks < 2; ++ks) {
      FragB16 pf;
      load_frag_lds(pf, sPw, 66, ks * 32, lane);
#pragma unroll
      for (int vt = 0; vt < 8; ++vt) {
        FragB16 vfr;
        load_frag_lds(vfr, sVt + vt * 16 * SV_STRIDE, SV_STRIDE, ks * 32, lane);
        o[vt] = wmma_bf16(pf, vfr, o[vt]);
      }
    }
  }

  // finalize: divide by row sum, write bf16 [S, H*hd]
  float inv[8];
#pragma unroll
  for (int r = 0; r < 8; ++r) inv[r] = 1.0f / lrow[r];
#pragma unroll
  for (int vt = 0; vt < 8; ++vt)
#pragma unroll
    for (int r = 0; r < 8; ++r) {
      const int row = qw0 + r + 8 * hi;
      O[(size_t)row * DMODEL + h * HD + vt * 16 + nn] = f2bf(o[vt][r] * inv[r]);
    }
}

extern "C" void kernel_launch(void* const* d_in, const int* in_sizes, int n_in,
                              void* d_out, int out_size, void* d_ws, size_t ws_size,
                              hipStream_t stream) {
  const float* hs = (const float*)d_in[0];
  // d_in[1] = attention_mask: unused (causality computed in-register)
  const int* pos = (const int*)d_in[2];
  const float* Wq = (const float*)d_in[3];
  const float* Wk = (const float*)d_in[4];
  const float* Wv = (const float*)d_in[5];
  const float* Wo = (const float*)d_in[6];
  const int S = in_sizes[0] / DMODEL;

  float* q = (float*)d_ws;                                 // [S, 2048] f32
  float* k = q + (size_t)S * DMODEL;                       // [S, 512]  f32
  float* v = k + (size_t)S * DKV;                          // [S, 512]  f32
  uint16_t* hsb = (uint16_t*)(v + (size_t)S * DKV);        // [S, 2048] bf16
  uint16_t* Wqb = hsb + (size_t)S * DMODEL;                // [2048, 2048] bf16
  uint16_t* Wkb = Wqb + (size_t)DMODEL * DMODEL;           // [512, 2048]  bf16
  uint16_t* Wvb = Wkb + (size_t)DKV * DMODEL;              // [512, 2048]  bf16
  uint16_t* Wob = Wvb + (size_t)DKV * DMODEL;              // [2048, 2048] bf16
  uint16_t* qb  = Wob + (size_t)DMODEL * DMODEL;           // [S, 2048] bf16 (scaled)
  uint16_t* kbf = qb + (size_t)S * DMODEL;                 // [S, 512]  bf16
  uint16_t* vtb = kbf + (size_t)S * DKV;                   // [512, S]  bf16 (V^T)
  uint16_t* aob = vtb + (size_t)S * DKV;                   // [S, 2048] bf16
  float* out = (float*)d_out;

  dim3 blk(256);
  const int nHS = S * DMODEL / 4, nWq = DMODEL * DMODEL / 4, nWk = DKV * DMODEL / 4;
  pack_bf16<<<(nHS + 255) / 256, blk, 0, stream>>>(hs, hsb, nHS);
  pack_bf16<<<(nWq + 255) / 256, blk, 0, stream>>>(Wq, Wqb, nWq);
  pack_bf16<<<(nWk + 255) / 256, blk, 0, stream>>>(Wk, Wkb, nWk);
  pack_bf16<<<(nWk + 255) / 256, blk, 0, stream>>>(Wv, Wvb, nWk);
  pack_bf16<<<(nWq + 255) / 256, blk, 0, stream>>>(Wo, Wob, nWq);

  gemm_abt_bf16<<<dim3(DMODEL / 64, S / 128), blk, 0, stream>>>(hsb, Wqb, q, S, DMODEL, DMODEL);
  gemm_abt_bf16<<<dim3(DKV / 64, S / 128), blk, 0, stream>>>(hsb, Wkb, k, S, DKV, DMODEL);
  gemm_abt_bf16<<<dim3(DKV / 64, S / 128), blk, 0, stream>>>(hsb, Wvb, v, S, DKV, DMODEL);

  const int rope_total = S * (H_HEADS + KV_HEADS) * (HD / 2);
  rope_kernel<<<(rope_total + 255) / 256, blk, 0, stream>>>(q, k, pos, S);

  const size_t conv_total = (size_t)S * DMODEL;
  convert_post_rope<<<(unsigned)((conv_total + 255) / 256), blk, 0, stream>>>(
      q, k, v, qb, kbf, vtb, S);

  attn_kernel<<<dim3(S / 128, H_HEADS), blk, 0, stream>>>(qb, kbf, vtb, aob, S);

  gemm_abt_bf16<<<dim3(DMODEL / 64, S / 128), blk, 0, stream>>>(aob, Wob, out, S, DMODEL, DMODEL);
}
